// SupplyChainTemporalGNN_43353399886391
// MI455X (gfx1250) — compile-verified, matrix operand order checked by reading, not run on hardware
//
#include <hip/hip_runtime.h>
#include <hip/hip_bf16.h>

// ---------------------------------------------------------------------------
// SupplyChainTemporalGNN forward on MI455X (gfx1250, wave32).
//  - All dense GEMMs via v_wmma_f32_16x16x32_f16, f32 accumulation.
//  - Weights prepacked once into B-fragment f16 layout -> 2x b128 loads/k-chunk.
//  - Multi-N-tile waves: A fragment loaded once per k-chunk, reused across
//    NT accumulators (cuts A operand traffic 4x on the dominant GEMMs).
//  - GAT edge softmax: wave32-per-edge streaming, float2 coalesced gathers,
//    shfl_xor reductions, atomic max via order-preserving u32 encoding.
// ---------------------------------------------------------------------------

#define Tz 8
#define Nz 10000
#define Ez 160000
#define Hz 64
#define ENz (Ez + Nz)     // edges + self loops  (multiple of 16)
#define TNz (Tz * Nz)     // multiple of 16
#define NEG_SLOPE 0.2f

typedef __attribute__((ext_vector_type(16))) _Float16 v16h;
typedef __attribute__((ext_vector_type(8)))  float    v8f;

// ---------------- order-preserving float <-> u32 for atomic max -------------
__device__ __forceinline__ unsigned fenc(float f) {
  unsigned u = __float_as_uint(f);
  return (u & 0x80000000u) ? ~u : (u | 0x80000000u);
}
__device__ __forceinline__ float fdec(unsigned u) {
  u = (u & 0x80000000u) ? (u & 0x7FFFFFFFu) : ~u;
  return __uint_as_float(u);
}

// ---------------------------------------------------------------------------
// Prepack weight W (f32, element (k,n) = W[k*wsk + n*wsn]) into f16 B-fragments.
// Layout: frag[((nt*ktiles + kt)*32 + lane)*16 + e]; zero pad k>=K / n>=Nout.
// ---------------------------------------------------------------------------
__global__ void pack_w(const float* __restrict__ W, int wsk, int wsn,
                       int K, int Nout, int ktiles, int ntiles,
                       _Float16* __restrict__ Bp) {
  int i = blockIdx.x * blockDim.x + threadIdx.x;
  int total = ntiles * ktiles * 32 * 16;
  if (i >= total) return;
  int e    = i & 15;
  int lane = (i >> 4) & 31;
  int rest = i >> 9;
  int kt   = rest % ktiles;
  int nt   = rest / ktiles;
  int half = lane >> 4, l16 = lane & 15;
  int k = kt * 32 + ((e < 8) ? (half * 8 + e) : (16 + half * 8 + (e - 8)));
  int n = nt * 16 + l16;
  float v = (k < K && n < Nout) ? W[(size_t)k * wsk + (size_t)n * wsn] : 0.0f;
  Bp[i] = (_Float16)v;
}

// ---------------------------------------------------------------------------
// WMMA GEMM: C[M x Nout] = act( A[M x Kp](f32, Kp%32==0, M%16==0) * Bp + bias )
// One wave32 per (16 x NT*16) tile strip; A fragment loaded once per k-chunk
// and reused across NT WMMA accumulators. Branch-free inner loop.
// act: 0 none, 1 relu. Requires ((Nout+15)/16) % NT == 0.
// ---------------------------------------------------------------------------
template <int NT>
__global__ void gemm_wmma_nt(const float* __restrict__ A,
                             const _Float16* __restrict__ Bp,
                             const float* __restrict__ bias,
                             float* __restrict__ C, int ldc,
                             int M, int Kp, int Nout, int act) {
  const int wave = threadIdx.x >> 5;
  const int lane = threadIdx.x & 31;
  const int ntiles  = (Nout + 15) >> 4;
  const int ngroups = ntiles / NT;
  const int ktiles  = Kp >> 5;
  const int tile = blockIdx.x * (blockDim.x >> 5) + wave;
  if (tile >= (M >> 4) * ngroups) return;       // wave-uniform exit
  const int mt  = tile / ngroups;
  const int ng  = tile - mt * ngroups;
  const int nt0 = ng * NT;
  const int half = lane >> 4;
  const int l16  = lane & 15;

  const float*    Arow = A + (size_t)(mt * 16 + l16) * Kp + half * 8;
  const _Float16* Bt0  = Bp + (size_t)nt0 * ktiles * 512 + (size_t)lane * 16;

  v8f acc[NT];
#pragma unroll
  for (int u = 0; u < NT; ++u) acc[u] = (v8f){};

  for (int kt = 0; kt < ktiles; ++kt) {
    const float* a0 = Arow + kt * 32;
    const float4 f0 = *(const float4*)(a0);
    const float4 f1 = *(const float4*)(a0 + 4);
    const float4 f2 = *(const float4*)(a0 + 16);
    const float4 f3 = *(const float4*)(a0 + 20);
    v16h a;
    a[0]  = (_Float16)f0.x; a[1]  = (_Float16)f0.y;
    a[2]  = (_Float16)f0.z; a[3]  = (_Float16)f0.w;
    a[4]  = (_Float16)f1.x; a[5]  = (_Float16)f1.y;
    a[6]  = (_Float16)f1.z; a[7]  = (_Float16)f1.w;
    a[8]  = (_Float16)f2.x; a[9]  = (_Float16)f2.y;
    a[10] = (_Float16)f2.z; a[11] = (_Float16)f2.w;
    a[12] = (_Float16)f3.x; a[13] = (_Float16)f3.y;
    a[14] = (_Float16)f3.z; a[15] = (_Float16)f3.w;
#pragma unroll
    for (int u = 0; u < NT; ++u) {
      const v16h b = *(const v16h*)(Bt0 + ((size_t)u * ktiles + kt) * 512);
      acc[u] = __builtin_amdgcn_wmma_f32_16x16x32_f16(
          false, a, false, b, (short)0, acc[u], false, false);
    }
  }

#pragma unroll
  for (int u = 0; u < NT; ++u) {
    const int bcol = (nt0 + u) * 16 + l16;
    if (bcol < Nout) {
      const float bv = bias ? bias[bcol] : 0.0f;
#pragma unroll
      for (int r = 0; r < 8; ++r) {
        float v = acc[u][r] + bv;
        if (act == 1) v = v > 0.0f ? v : 0.0f;
        C[(size_t)(mt * 16 + r + 8 * half) * ldc + bcol] = v;
      }
    }
  }
}

// ---------------- tiny utility kernels -------------------------------------
__global__ void fill_f32(float* p, float v, long long n) {
  long long i = blockIdx.x * (long long)blockDim.x + threadIdx.x;
  if (i < n) p[i] = v;
}
__global__ void fill_u32(unsigned* p, unsigned v, int n) {
  int i = blockIdx.x * blockDim.x + threadIdx.x;
  if (i < n) p[i] = v;
}
__global__ void bias_bcast64(float* p, const float* __restrict__ bias, long long n) {
  long long i = blockIdx.x * (long long)blockDim.x + threadIdx.x;
  if (i < n) p[i] = bias[i & 63];
}
__global__ void copy_f32(float* __restrict__ dst, const float* __restrict__ src, int n) {
  int i = blockIdx.x * blockDim.x + threadIdx.x;
  if (i < n) dst[i] = src[i];
}
// copy [M x K] into zero-padded [M x Kp]
__global__ void pad_copy(const float* __restrict__ src, int K,
                         float* __restrict__ dst, int Kp, long long M) {
  long long i = blockIdx.x * (long long)blockDim.x + threadIdx.x;
  if (i >= M * Kp) return;
  int k = (int)(i % Kp);
  long long r = i / Kp;
  dst[i] = (k < K) ? src[r * K + k] : 0.0f;
}

// build src_f / dst_f (edges then self-loops), int32
__global__ void build_edges(const int* __restrict__ eidx, int* srcf, int* dstf) {
  int i = blockIdx.x * blockDim.x + threadIdx.x;
  if (i >= ENz) return;
  if (i < Ez) { srcf[i] = eidx[i]; dstf[i] = eidx[Ez + i]; }
  else        { srcf[i] = i - Ez;  dstf[i] = i - Ez; }
}

__global__ void deg_count(const int* __restrict__ dstf, float* deg) {
  int i = blockIdx.x * blockDim.x + threadIdx.x;
  if (i < Ez) atomicAdd(&deg[dstf[i]], 1.0f);
}

// loop_attr: ep rows [Ez,ENz) (pitch 32) += segment_sum(ep[0:Ez], dst)
__global__ void loop_accum(const float* ep, const int* __restrict__ dstf, float* ep_loop) {
  long long i = blockIdx.x * (long long)blockDim.x + threadIdx.x;
  if (i >= (long long)Ez * 8) return;
  int e = (int)(i >> 3), f = (int)(i & 7);
  atomicAdd(&ep_loop[(size_t)dstf[e] * 32 + f], ep[(size_t)e * 32 + f]);
}
__global__ void loop_div(float* ep_loop, const float* __restrict__ deg) {
  int i = blockIdx.x * blockDim.x + threadIdx.x;
  if (i >= Nz * 8) return;
  ep_loop[(i >> 3) * 32 + (i & 7)] /= fmaxf(deg[i >> 3], 1.0f);
}

// ------------- GAT: one wave32 per edge, float2 coalesced, shfl reduce ------
__global__ void gat_score(const int* __restrict__ srcf, const int* __restrict__ dstf,
                          const float* __restrict__ xl_t, const float* __restrict__ xr_t,
                          const float* __restrict__ e_emb, const float* __restrict__ att,
                          float* __restrict__ a_out, unsigned* amax_u) {
  const int wave = threadIdx.x >> 5, lane = threadIdx.x & 31;
  const int e = blockIdx.x * (blockDim.x >> 5) + wave;
  if (e >= ENz) return;
  const int s = srcf[e], d = dstf[e];
  const int j = lane * 2;
  const float2 vl = *(const float2*)(xl_t + (size_t)s * Hz + j);
  const float2 vr = *(const float2*)(xr_t + (size_t)d * Hz + j);
  const float2 ve = *(const float2*)(e_emb + (size_t)e * Hz + j);
  const float2 va = *(const float2*)(att + j);
  float s0 = vl.x + vr.x + ve.x;
  float s1 = vl.y + vr.y + ve.y;
  s0 = s0 > 0.0f ? s0 : NEG_SLOPE * s0;
  s1 = s1 > 0.0f ? s1 : NEG_SLOPE * s1;
  float p = s0 * va.x + s1 * va.y;
#pragma unroll
  for (int m = 16; m >= 1; m >>= 1) p += __shfl_xor(p, m, 32);
  if (lane == 0) {
    a_out[e] = p;
    atomicMax(&amax_u[d], fenc(p));
  }
}

__global__ void gat_exp(const float* __restrict__ a, const int* __restrict__ dstf,
                        const unsigned* __restrict__ amax_u,
                        float* __restrict__ ex, float* denom) {
  int e = blockIdx.x * blockDim.x + threadIdx.x;
  if (e >= ENz) return;
  int d = dstf[e];
  float v = __expf(a[e] - fdec(amax_u[d]));
  ex[e] = v;
  atomicAdd(&denom[d], v);
}

__global__ void gat_aggr(const int* __restrict__ srcf, const int* __restrict__ dstf,
                         const float* __restrict__ ex, const float* __restrict__ denom,
                         const float* __restrict__ xl_t, float* g_t) {
  const int wave = threadIdx.x >> 5, lane = threadIdx.x & 31;
  const int e = blockIdx.x * (blockDim.x >> 5) + wave;
  if (e >= ENz) return;
  const int s = srcf[e], d = dstf[e];
  const float alpha = ex[e] / denom[d];
  const int j = lane * 2;
  const float2 vl = *(const float2*)(xl_t + (size_t)s * Hz + j);
  atomicAdd(&g_t[(size_t)d * Hz + j],     alpha * vl.x);
  atomicAdd(&g_t[(size_t)d * Hz + j + 1], alpha * vl.y);
}

// ------------------------------- GRU pointwise ------------------------------
__global__ void gru_point(const float* __restrict__ gi, const float* __restrict__ gh,
                          float* hstate, float* __restrict__ hout_t) {
  int i = blockIdx.x * blockDim.x + threadIdx.x;
  if (i >= Nz * Hz) return;
  int n = i >> 6, j = i & 63;
  const size_t b = (size_t)n * 192 + j;
  float r  = 1.0f / (1.0f + __expf(-(gi[b]       + gh[b])));
  float z  = 1.0f / (1.0f + __expf(-(gi[b + 64]  + gh[b + 64])));
  float nn = tanhf(gi[b + 128] + r * gh[b + 128]);
  float h  = hstate[i];
  float hn = (1.0f - z) * nn + z * h;
  hstate[i] = hn;
  hout_t[i] = hn;
}

// ----------------------- head second layer: dot(hid, W2) --------------------
__global__ void head2(const float* __restrict__ hid, const float* __restrict__ W2,
                      const float* __restrict__ b2, float* __restrict__ out) {
  const int wave = threadIdx.x >> 5, lane = threadIdx.x & 31;
  const int n = blockIdx.x * (blockDim.x >> 5) + wave;
  if (n >= Nz) return;
  float p = hid[(size_t)n * 32 + lane] * W2[lane];
#pragma unroll
  for (int m = 16; m >= 1; m >>= 1) p += __shfl_xor(p, m, 32);
  if (lane == 0) out[n] = p + b2[0];
}

// ---------------------------------------------------------------------------
extern "C" void kernel_launch(void* const* d_in, const int* in_sizes, int n_in,
                              void* d_out, int out_size, void* d_ws, size_t ws_size,
                              hipStream_t stream) {
  (void)in_sizes; (void)n_in; (void)out_size; (void)ws_size;

  const float* node_features = (const float*)d_in[0];   // (1,8,10000,16)
  const int*   edge_index    = (const int*)  d_in[1];   // (2,160000)
  const float* edge_attr     = (const float*)d_in[2];   // (160000,8)
  const float* enc_W = (const float*)d_in[3];
  const float* enc_b = (const float*)d_in[4];
  const float* edge_enc_W = (const float*)d_in[5];
  const float* edge_enc_b = (const float*)d_in[6];
  const int LBASE[2] = {7, 20};   // proj_W,proj_b,Wl,bl,Wr,br,Wedge,att,bias,Wih,Whh,bih,bhh
  const float* order_W1  = (const float*)d_in[33];
  const float* order_b1  = (const float*)d_in[34];
  const float* order_W2  = (const float*)d_in[35];
  const float* order_b2  = (const float*)d_in[36];
  const float* demand_W1 = (const float*)d_in[37];
  const float* demand_b1 = (const float*)d_in[38];
  const float* demand_W2 = (const float*)d_in[39];
  const float* demand_b2 = (const float*)d_in[40];

  // ---- workspace carve
  char* wptr = (char*)d_ws;
  auto alloc = [&](size_t bytes) -> void* {
    void* p = (void*)wptr;
    wptr += (bytes + 255) & ~(size_t)255;
    return p;
  };
  int*      srcf   = (int*)alloc(sizeof(int) * ENz);
  int*      dstf   = (int*)alloc(sizeof(int) * ENz);
  float*    deg    = (float*)alloc(sizeof(float) * Nz);
  float*    nf_pad = (float*)alloc(sizeof(float) * (size_t)TNz * 32);
  float*    ea_pad = (float*)alloc(sizeof(float) * (size_t)Ez * 32);
  float*    hA     = (float*)alloc(sizeof(float) * (size_t)TNz * Hz);
  float*    hB     = (float*)alloc(sizeof(float) * (size_t)TNz * Hz);
  float*    e_embd = (float*)alloc(sizeof(float) * (size_t)Ez * Hz);
  float*    ep     = (float*)alloc(sizeof(float) * (size_t)ENz * 32); // pitch 32, K-padded
  float*    ee     = (float*)alloc(sizeof(float) * (size_t)ENz * Hz);
  float*    xl     = (float*)alloc(sizeof(float) * (size_t)TNz * Hz);
  float*    xr     = (float*)alloc(sizeof(float) * (size_t)TNz * Hz);
  float*    a_buf  = (float*)alloc(sizeof(float) * ENz);
  float*    ex_buf = (float*)alloc(sizeof(float) * ENz);
  unsigned* amax_u = (unsigned*)alloc(sizeof(unsigned) * Nz);
  float*    denom  = (float*)alloc(sizeof(float) * Nz);
  float*    g_buf  = (float*)alloc(sizeof(float) * (size_t)TNz * Hz);
  float*    hstate = (float*)alloc(sizeof(float) * (size_t)Nz * Hz);
  float*    gi     = (float*)alloc(sizeof(float) * (size_t)Nz * 192);
  float*    gh     = (float*)alloc(sizeof(float) * (size_t)Nz * 192);
  float*    hid    = (float*)alloc(sizeof(float) * (size_t)Nz * 32);
  // packed f16 weights (B-fragments)
  _Float16* P_enc   = (_Float16*)alloc(sizeof(_Float16) * 1 * 4 * 512);
  _Float16* P_eenc  = (_Float16*)alloc(sizeof(_Float16) * 1 * 4 * 512);
  _Float16* P_proj[2]; _Float16* P_wedge[2];
  _Float16* P_wl[2];   _Float16* P_wr[2];
  _Float16* P_wih[2];  _Float16* P_whh[2];
  for (int l = 0; l < 2; ++l) {
    P_proj[l]  = (_Float16*)alloc(sizeof(_Float16) * 2 * 1 * 512);
    P_wedge[l] = (_Float16*)alloc(sizeof(_Float16) * 1 * 4 * 512);
    P_wl[l]    = (_Float16*)alloc(sizeof(_Float16) * 2 * 4 * 512);
    P_wr[l]    = (_Float16*)alloc(sizeof(_Float16) * 2 * 4 * 512);
    P_wih[l]   = (_Float16*)alloc(sizeof(_Float16) * 2 * 12 * 512);
    P_whh[l]   = (_Float16*)alloc(sizeof(_Float16) * 2 * 12 * 512);
  }
  _Float16* P_oW1 = (_Float16*)alloc(sizeof(_Float16) * 2 * 2 * 512);
  _Float16* P_dW1 = (_Float16*)alloc(sizeof(_Float16) * 2 * 2 * 512);

  const int TPB = 256;
  auto cdiv = [](long long a, long long b) { return (int)((a + b - 1) / b); };
  auto pack = [&](const float* W, int wsk, int wsn, int K, int Nout,
                  int ktiles, int ntiles, _Float16* Bp) {
    int tot = ntiles * ktiles * 512;
    pack_w<<<cdiv(tot, TPB), TPB, 0, stream>>>(W, wsk, wsn, K, Nout, ktiles, ntiles, Bp);
  };
  // dispatch: widest NT that divides the N-tile count (A-fragment reuse)
  auto gemm = [&](const float* A, const _Float16* Bp, const float* bias,
                  float* C, int ldc, int M, int Kp, int Nout, int act) {
    int ntiles = (Nout + 15) >> 4;
    if ((ntiles & 3) == 0) {
      int tiles = (M >> 4) * (ntiles >> 2);
      gemm_wmma_nt<4><<<cdiv(tiles, 8), TPB, 0, stream>>>(A, Bp, bias, C, ldc, M, Kp, Nout, act);
    } else if ((ntiles & 1) == 0) {
      int tiles = (M >> 4) * (ntiles >> 1);
      gemm_wmma_nt<2><<<cdiv(tiles, 8), TPB, 0, stream>>>(A, Bp, bias, C, ldc, M, Kp, Nout, act);
    } else {
      int tiles = (M >> 4) * ntiles;
      gemm_wmma_nt<1><<<cdiv(tiles, 8), TPB, 0, stream>>>(A, Bp, bias, C, ldc, M, Kp, Nout, act);
    }
  };

  // ---- prepack all weights (tiny; resident in L2 for the whole launch)
  pack(enc_W,      64, 1, 16, 64, 1, 4,  P_enc);
  pack(edge_enc_W, 64, 1,  8, 64, 1, 4,  P_eenc);
  for (int l = 0; l < 2; ++l) {
    const int b = LBASE[l];
    pack((const float*)d_in[b + 0],  8, 1, 64,   8, 2, 1,  P_proj[l]);   // proj_W
    pack((const float*)d_in[b + 6], 64, 1,  8,  64, 1, 4,  P_wedge[l]);  // Wedge
    pack((const float*)d_in[b + 2], 64, 1, 64,  64, 2, 4,  P_wl[l]);     // Wl
    pack((const float*)d_in[b + 4], 64, 1, 64,  64, 2, 4,  P_wr[l]);     // Wr
    pack((const float*)d_in[b + 9],  1, 64, 64, 192, 2, 12, P_wih[l]);   // Wih^T
    pack((const float*)d_in[b +10],  1, 64, 64, 192, 2, 12, P_whh[l]);   // Whh^T
  }
  pack(order_W1,  32, 1, 64, 32, 2, 2, P_oW1);
  pack(demand_W1, 32, 1, 64, 32, 2, 2, P_dW1);

  // ---- static graph prep + padded activation inputs
  build_edges<<<cdiv(ENz, TPB), TPB, 0, stream>>>(edge_index, srcf, dstf);
  fill_f32<<<cdiv(Nz, TPB), TPB, 0, stream>>>(deg, 0.0f, Nz);
  deg_count<<<cdiv(Ez, TPB), TPB, 0, stream>>>(dstf, deg);
  pad_copy<<<cdiv((long long)TNz * 32, TPB), TPB, 0, stream>>>(node_features, 16, nf_pad, 32, TNz);
  pad_copy<<<cdiv((long long)Ez * 32, TPB), TPB, 0, stream>>>(edge_attr, 8, ea_pad, 32, Ez);

  // ---- encoders
  gemm(nf_pad, P_enc,  enc_b,      hA,     Hz, TNz, 32, Hz, 0);
  gemm(ea_pad, P_eenc, edge_enc_b, e_embd, Hz, Ez,  32, Hz, 0);

  float* hcur = hA;
  float* hnext = hB;

  for (int l = 0; l < 2; ++l) {
    const int b = LBASE[l];
    const float* proj_b = (const float*)d_in[b + 1];
    const float* bl     = (const float*)d_in[b + 3];
    const float* br     = (const float*)d_in[b + 5];
    const float* att    = (const float*)d_in[b + 7];
    const float* gbias  = (const float*)d_in[b + 8];
    const float* bih    = (const float*)d_in[b + 11];
    const float* bhh    = (const float*)d_in[b + 12];

    // ep = edge_emb @ proj_W + proj_b (E x 8), pitch 32 with zeroed pad;
    // rows [Ez,ENz) become loop_attr accumulators (also zeroed here).
    fill_f32<<<cdiv((long long)ENz * 32, TPB), TPB, 0, stream>>>(ep, 0.0f, (long long)ENz * 32);
    gemm(e_embd, P_proj[l], proj_b, ep, 32, Ez, 64, 8, 0);
    loop_accum<<<cdiv((long long)Ez * 8, TPB), TPB, 0, stream>>>(ep, dstf, ep + (size_t)Ez * 32);
    loop_div<<<cdiv(Nz * 8, TPB), TPB, 0, stream>>>(ep + (size_t)Ez * 32, deg);

    // e_emb = concat(ep, loop_attr) @ Wedge   ((E+N) x 64)
    gemm(ep, P_wedge[l], nullptr, ee, Hz, ENz, 32, Hz, 0);

    // xl / xr projections over all timesteps at once
    gemm(hcur, P_wl[l], bl, xl, Hz, TNz, 64, Hz, 0);
    gemm(hcur, P_wr[l], br, xr, Hz, TNz, 64, Hz, 0);

    // g initialized to GAT bias; GRU state zeroed
    bias_bcast64<<<cdiv((long long)TNz * Hz, TPB), TPB, 0, stream>>>(g_buf, gbias, (long long)TNz * Hz);
    fill_f32<<<cdiv(Nz * Hz, TPB), TPB, 0, stream>>>(hstate, 0.0f, Nz * Hz);

    for (int t = 0; t < Tz; ++t) {
      const float* xl_t = xl + (size_t)t * Nz * Hz;
      const float* xr_t = xr + (size_t)t * Nz * Hz;
      float* g_t = g_buf + (size_t)t * Nz * Hz;

      fill_u32<<<cdiv(Nz, TPB), TPB, 0, stream>>>(amax_u, 0u, Nz);
      fill_f32<<<cdiv(Nz, TPB), TPB, 0, stream>>>(denom, 0.0f, Nz);
      gat_score<<<cdiv(ENz, 8), TPB, 0, stream>>>(srcf, dstf, xl_t, xr_t, ee, att, a_buf, amax_u);
      gat_exp<<<cdiv(ENz, TPB), TPB, 0, stream>>>(a_buf, dstf, amax_u, ex_buf, denom);
      gat_aggr<<<cdiv(ENz, 8), TPB, 0, stream>>>(srcf, dstf, ex_buf, denom, xl_t, g_t);

      // GRU step
      gemm(g_t,    P_wih[l], bih, gi, 192, Nz, 64, 192, 0);
      gemm(hstate, P_whh[l], bhh, gh, 192, Nz, 64, 192, 0);
      gru_point<<<cdiv(Nz * Hz, TPB), TPB, 0, stream>>>(gi, gh, hstate,
                                                        hnext + (size_t)t * Nz * Hz);
    }
    float* tmp = hcur; hcur = hnext; hnext = tmp;
  }

  // ---- heads on last timestep
  const float* last = hcur + (size_t)(Tz - 1) * Nz * Hz;
  float* out = (float*)d_out;

  gemm(last, P_oW1, order_b1, hid, 32, Nz, 64, 32, 1);
  head2<<<cdiv(Nz, 8), TPB, 0, stream>>>(hid, order_W2, order_b2, out);

  gemm(last, P_dW1, demand_b1, hid, 32, Nz, 64, 32, 1);
  head2<<<cdiv(Nz, 8), TPB, 0, stream>>>(hid, demand_W2, demand_b2, out + Nz);

  copy_f32<<<cdiv(Nz * Hz, TPB), TPB, 0, stream>>>(out + 2 * Nz, last, Nz * Hz);
}
